// Qwen35LinearAttention_12146167513340
// MI455X (gfx1250) — matrix operand order repaired and checked by reading, not compiled, not run on hardware
//
#include <hip/hip_runtime.h>
#include <math.h>

// ---------------------------------------------------------------------------
// Types for CDNA5 WMMA (wave32, v_wmma_f32_16x16x32_bf16)
// ---------------------------------------------------------------------------
typedef __bf16 bf16;
typedef bf16  v16bf __attribute__((ext_vector_type(16)));
typedef float v8f   __attribute__((ext_vector_type(8)));

__device__ __forceinline__ v8f v8f_zero() {
  v8f z;
#pragma unroll
  for (int i = 0; i < 8; ++i) z[i] = 0.0f;
  return z;
}

// A-operand fragment: 16x32 bf16 tile, LDS row-major [row][k], per ISA layout:
// lane L (half=L/16, m=L%16): elems 0..7 -> K = 8*half + e ; elems 8..15 -> K = 8*half + 16 + e
__device__ __forceinline__ v16bf frag_a_lds(const bf16* buf, int stride, int rbase, int kbase) {
  int lane = threadIdx.x & 31;
  int half = lane >> 4;
  const bf16* p = buf + (size_t)(rbase + (lane & 15)) * stride + kbase + 8 * half;
  v16bf a;
#pragma unroll
  for (int e = 0; e < 8; ++e) a[e] = p[e];
#pragma unroll
  for (int e = 0; e < 8; ++e) a[8 + e] = p[16 + e];
  return a;
}

// B-operand fragment: 32x16 bf16 tile; LDS stores B column-rows [n][k]:
// lane L: n = L%16, elems 0..7 -> K = 8*half + e ; elems 8..15 -> K = 8*half + 16 + e
__device__ __forceinline__ v16bf frag_b_lds(const bf16* buf, int stride, int nbase, int kbase) {
  int lane = threadIdx.x & 31;
  int half = lane >> 4;
  const bf16* p = buf + (size_t)(nbase + (lane & 15)) * stride + kbase + 8 * half;
  v16bf b;
#pragma unroll
  for (int e = 0; e < 8; ++e) b[e] = p[e];
#pragma unroll
  for (int e = 0; e < 8; ++e) b[8 + e] = p[16 + e];
  return b;
}

__device__ __forceinline__ v8f wmma_bf16(v16bf a, v16bf b, v8f c) {
  return __builtin_amdgcn_wmma_f32_16x16x32_bf16(false, a, false, b, (short)0, c, false, false);
}

__device__ __forceinline__ float sigmoidf_(float x) { return 1.0f / (1.0f + __expf(-x)); }
__device__ __forceinline__ float siluf_(float x) { return x * sigmoidf_(x); }

// ---------------------------------------------------------------------------
// Generic GEMM: Y[M,N] = X[M,K] @ W[N,K]^T   (fp32 in/out, bf16 WMMA compute)
// WG tile 128(M) x 128(N), 8 waves in 4x2: wave (wr=w>>1, wc=w&1) computes
// rows 32*wr..+31, cols 64*wc..+63 (2 A-frags x 4 B-frags = 8 WMMA / K-step).
// Double-buffered LDS, one barrier per K-step, prefetch into registers.
// grid: (ceil(N/128), M/128), block 256.
// ---------------------------------------------------------------------------
__global__ __launch_bounds__(256) void gemm_bf16_kernel(
    const float* __restrict__ X, const float* __restrict__ W, float* __restrict__ Y,
    int M, int N, int K) {
  __shared__ bf16 Ab[2][128 * 40];
  __shared__ bf16 Bb[2][128 * 40];
  const int tid = threadIdx.x;
  const int wave = tid >> 5;
  const int lane = tid & 31;
  const int half = lane >> 4;
  const int l16 = lane & 15;
  const int wr = wave >> 1;          // 0..3
  const int wc = wave & 1;           // 0..1
  const int row0 = blockIdx.y * 128;
  const int col0 = blockIdx.x * 128;

  v8f acc[2][4];
#pragma unroll
  for (int ia = 0; ia < 2; ++ia)
#pragma unroll
    for (int j = 0; j < 4; ++j) acc[ia][j] = v8f_zero();

  const int ar = tid >> 3;           // 0..31
  const int ac = (tid & 7) * 4;      // 0,4,..,28

  float4 ra[4], rb[4];

  // prefetch K-tile k0 into registers (A: 128x32, B: 128x32)
  auto load_tile = [&](int k0) {
#pragma unroll
    for (int rr = 0; rr < 4; ++rr) {
      int r = ar + rr * 32;
      ra[rr] = *(const float4*)&X[(size_t)(row0 + r) * K + k0 + ac];
    }
#pragma unroll
    for (int rr = 0; rr < 4; ++rr) {
      int nrow = col0 + ar + rr * 32;
      rb[rr] = make_float4(0.f, 0.f, 0.f, 0.f);
      if (nrow < N) rb[rr] = *(const float4*)&W[(size_t)nrow * K + k0 + ac];
    }
  };
  // convert + commit registers into LDS buffer b
  auto store_tile = [&](int b) {
#pragma unroll
    for (int rr = 0; rr < 4; ++rr) {
      int r = ar + rr * 32;
      bf16* d = &Ab[b][r * 40 + ac];
      d[0] = (bf16)ra[rr].x; d[1] = (bf16)ra[rr].y;
      d[2] = (bf16)ra[rr].z; d[3] = (bf16)ra[rr].w;
    }
#pragma unroll
    for (int rr = 0; rr < 4; ++rr) {
      int r = ar + rr * 32;
      bf16* d = &Bb[b][r * 40 + ac];
      d[0] = (bf16)rb[rr].x; d[1] = (bf16)rb[rr].y;
      d[2] = (bf16)rb[rr].z; d[3] = (bf16)rb[rr].w;
    }
  };

  load_tile(0);
  store_tile(0);
  __syncthreads();

  const int nk = K / 32;
  for (int kb = 0; kb < nk; ++kb) {
    const int cur = kb & 1, nxt = cur ^ 1;
    if (kb + 1 < nk) load_tile((kb + 1) * 32);  // loads in flight during WMMA

    v16bf a0 = frag_a_lds(Ab[cur], 40, 32 * wr, 0);
    v16bf a1 = frag_a_lds(Ab[cur], 40, 32 * wr + 16, 0);
#pragma unroll
    for (int j = 0; j < 4; ++j) {
      v16bf bfg = frag_b_lds(Bb[cur], 40, 64 * wc + 16 * j, 0);
      acc[0][j] = wmma_bf16(a0, bfg, acc[0][j]);
      acc[1][j] = wmma_bf16(a1, bfg, acc[1][j]);
    }

    if (kb + 1 < nk) store_tile(nxt);
    __syncthreads();
  }

#pragma unroll
  for (int ia = 0; ia < 2; ++ia) {
    const int r0 = row0 + 32 * wr + 16 * ia + 8 * half;
#pragma unroll
    for (int j = 0; j < 4; ++j) {
      int c = col0 + 64 * wc + 16 * j + l16;
      if (c < N) {
#pragma unroll
        for (int i = 0; i < 8; ++i) Y[(size_t)(r0 + i) * N + c] = acc[ia][j][i];
      }
    }
  }
}

// ---------------------------------------------------------------------------
// Depthwise causal conv (K=4) + SiLU.  grid: (C/256, S)
// ---------------------------------------------------------------------------
__global__ __launch_bounds__(256) void conv_silu_kernel(
    const float* __restrict__ qkv_raw, const float* __restrict__ conv_w,
    float* __restrict__ qkv_act, int S, int C) {
  const int c = blockIdx.x * 256 + threadIdx.x;
  const int s = blockIdx.y;
  const float w0 = conv_w[c * 4 + 0], w1 = conv_w[c * 4 + 1];
  const float w2 = conv_w[c * 4 + 2], w3 = conv_w[c * 4 + 3];
  float acc = w3 * qkv_raw[(size_t)s * C + c];
  if (s >= 1) acc += w2 * qkv_raw[(size_t)(s - 1) * C + c];
  if (s >= 2) acc += w1 * qkv_raw[(size_t)(s - 2) * C + c];
  if (s >= 3) acc += w0 * qkv_raw[(size_t)(s - 3) * C + c];
  qkv_act[(size_t)s * C + c] = siluf_(acc);
}

// ---------------------------------------------------------------------------
// q/k L2-norm + pack bf16 (q scaled by DK^-0.5). 1 wave per (s, head, q|k).
// grid: S*32/8 blocks of 256.
// ---------------------------------------------------------------------------
__global__ __launch_bounds__(256) void qknorm_kernel(
    const float* __restrict__ qkv_act, bf16* __restrict__ q_bf, bf16* __restrict__ k_bf) {
  const int wave = threadIdx.x >> 5, lane = threadIdx.x & 31;
  const int task = blockIdx.x * 8 + wave;   // s*32 + unit
  const int s = task >> 5, u = task & 31;
  const bool isK = (u >= 16);
  const int h = u & 15;
  const float* src = qkv_act + (size_t)s * 8192 + (isK ? 2048 : 0) + h * 128;
  float v[4];
  float ss = 0.0f;
#pragma unroll
  for (int i = 0; i < 4; ++i) { v[i] = src[lane + 32 * i]; ss += v[i] * v[i]; }
#pragma unroll
  for (int m = 16; m >= 1; m >>= 1) ss += __shfl_xor(ss, m, 32);
  float sc = rsqrtf(ss + 1e-6f) * (isK ? 1.0f : 0.08838834764831845f);  // 128^-0.5
  bf16* dst = (isK ? k_bf : q_bf) + ((size_t)s * 16 + h) * 128;
#pragma unroll
  for (int i = 0; i < 4; ++i) dst[lane + 32 * i] = (bf16)(v[i] * sc);
}

// v pack to bf16 ([S,32,128] contiguous). grid: S*4096/256
__global__ __launch_bounds__(256) void vpack_kernel(
    const float* __restrict__ qkv_act, bf16* __restrict__ v_bf) {
  const size_t idx = (size_t)blockIdx.x * 256 + threadIdx.x;
  const size_t s = idx >> 12;          // /4096
  const size_t c = idx & 4095;
  v_bf[idx] = (bf16)qkv_act[s * 8192 + 4096 + c];
}

// beta = sigmoid, g = -exp(A_log)*softplus(a+dt_bias). grid: S*32/256
__global__ __launch_bounds__(256) void betag_kernel(
    const float* __restrict__ braw, const float* __restrict__ araw,
    const float* __restrict__ dt_bias, const float* __restrict__ A_log,
    float* __restrict__ beta, float* __restrict__ g) {
  const int idx = blockIdx.x * 256 + threadIdx.x;
  const int h = idx & 31;
  beta[idx] = sigmoidf_(braw[idx]);
  float x = araw[idx] + dt_bias[h];
  float sp = (x > 20.0f) ? x : log1pf(__expf(x));
  g[idx] = -__expf(A_log[h]) * sp;
}

// per-chunk cumulative sum of g. one thread per (chunk, head). grid: 2048/256
__global__ __launch_bounds__(256) void gcum_kernel(
    const float* __restrict__ g, float* __restrict__ gc) {
  const int idx = blockIdx.x * 256 + threadIdx.x;  // 0..2047
  const int n = idx >> 5, h = idx & 31;
  float run = 0.0f;
  for (int t = 0; t < 64; ++t) {
    run += g[((size_t)(n * 64 + t)) * 32 + h];
    gc[((size_t)(n * 64 + t)) * 32 + h] = run;
  }
}

// ---------------------------------------------------------------------------
// Intra-chunk kernel: per (chunk n, head h):
//   M = tril_strict((k_beta @ k^T) * exp(gc_i - gc_j));  T = (I+M)^-1
//   v_t = T @ v_beta ;  k_cum = T @ (k_beta * exp(gc))
// grid: (64, 32), block 256 (8 waves)
// ---------------------------------------------------------------------------
__global__ __launch_bounds__(256) void chunk_prep_kernel(
    const bf16* __restrict__ k_bf, const bf16* __restrict__ v_bf,
    const float* __restrict__ beta, const float* __restrict__ gc,
    bf16* __restrict__ vt_bf, bf16* __restrict__ kc_bf) {
  __shared__ bf16 SKB[64 * 136];    // k_beta, row-major [t][d]
  __shared__ bf16 SK[64 * 136];     // k       row-major [t][d]
  __shared__ bf16 SVBT[128 * 72];   // v_beta^T [d][t]
  __shared__ bf16 SKGT[128 * 72];   // (k_beta*exp(gc))^T [d][t]
  __shared__ float MA[64 * 68];     // -M
  __shared__ float TF[64 * 68];     // T (fp32)
  __shared__ bf16 TB[64 * 72];      // T (bf16)
  __shared__ float GCL[64];

  const int n = blockIdx.x, h = blockIdx.y, hk = h >> 1;
  const int s0 = n * 64;
  const int tid = threadIdx.x, wave = tid >> 5, lane = tid & 31, half = lane >> 4;

  {  // stage
    const int r = tid >> 2, c0 = (tid & 3) * 32;
    const float br = beta[(size_t)(s0 + r) * 32 + h];
    const float eg = __expf(gc[(size_t)(s0 + r) * 32 + h]);
    const bf16* krow = k_bf + ((size_t)(s0 + r) * 16 + hk) * 128;
    const bf16* vrow = v_bf + ((size_t)(s0 + r) * 32 + h) * 128;
    for (int c = c0; c < c0 + 32; ++c) {
      float kv = (float)krow[c];
      SK[r * 136 + c] = (bf16)kv;
      float kb = kv * br;
      SKB[r * 136 + c] = (bf16)kb;
      SKGT[c * 72 + r] = (bf16)(kb * eg);
      SVBT[c * 72 + r] = (bf16)((float)vrow[c] * br);
    }
    if (tid < 64) GCL[tid] = gc[(size_t)(s0 + tid) * 32 + h];
  }
  __syncthreads();

  // MA = -tril_strict((kb @ k^T) * decay)
#pragma unroll
  for (int p = 0; p < 2; ++p) {
    int id = wave * 2 + p;
    int mt = id >> 2, nt = id & 3;
    v8f c = v8f_zero();
#pragma unroll
    for (int kk = 0; kk < 128; kk += 32) {
      v16bf a = frag_a_lds(SKB, 136, mt * 16, kk);
      v16bf b = frag_b_lds(SK, 136, nt * 16, kk);
      c = wmma_bf16(a, b, c);
    }
    int j = nt * 16 + (lane & 15);
#pragma unroll
    for (int i8 = 0; i8 < 8; ++i8) {
      int i = mt * 16 + i8 + 8 * half;
      MA[i * 68 + j] = (i > j) ? -c[i8] * __expf(GCL[i] - GCL[j]) : 0.0f;
    }
  }
  __syncthreads();

  // forward substitution: row i of T = e_i + sum_{j<i} MA[i][j] * T[j]
  for (int i = 0; i < 64; ++i) {
    if (tid < 64) {
      const int cc = tid;
      float acc = 0.0f;
      if (cc <= i) {
        acc = (cc == i) ? 1.0f : 0.0f;
        for (int j = cc; j < i; ++j) acc += MA[i * 68 + j] * TF[j * 68 + cc];
      }
      TF[i * 68 + cc] = acc;
    }
    __syncthreads();
  }
  for (int e = tid; e < 64 * 64; e += 256) {
    int i = e >> 6, j = e & 63;
    TB[i * 72 + j] = (bf16)TF[i * 68 + j];
  }
  __syncthreads();

  // v_t / k_cum = T @ RHS  (RHS stored transposed as B operand)
#pragma unroll
  for (int p = 0; p < 8; ++p) {
    int id = wave * 8 + p;            // 0..63
    int sel = id >> 5;                // 0: v_t  1: k_cum
    int rem = id & 31;
    int mt = rem >> 3, nt = rem & 7;
    v8f c = v8f_zero();
#pragma unroll
    for (int kk = 0; kk < 64; kk += 32) {
      v16bf a = frag_a_lds(TB, 72, mt * 16, kk);
      v16bf b = frag_b_lds(sel ? SKGT : SVBT, 72, nt * 16, kk);
      c = wmma_bf16(a, b, c);
    }
    bf16* dst = (sel ? kc_bf : vt_bf) + ((size_t)h * 64 + n) * 64 * 128;
    int d = nt * 16 + (lane & 15);
#pragma unroll
    for (int i8 = 0; i8 < 8; ++i8) {
      int t = mt * 16 + i8 + 8 * half;
      dst[(size_t)t * 128 + d] = (bf16)c[i8];
    }
  }
}

// ---------------------------------------------------------------------------
// Sequential inter-chunk scan, one workgroup per head (32 WGs, 8 waves).
// state[128x128] lives in WMMA accumulators across the 64-chunk loop.
// ---------------------------------------------------------------------------
__global__ __launch_bounds__(256) void scan_kernel(
    const bf16* __restrict__ q_bf, const bf16* __restrict__ k_bf,
    const float* __restrict__ gc,
    const bf16* __restrict__ vt_bf, const bf16* __restrict__ kc_bf,
    float* __restrict__ core) {
  __shared__ bf16 SQ[64 * 136];     // q (phase3: scaled by exp(gc))
  __shared__ bf16 SKK[64 * 136];    // k
  __shared__ bf16 SKC[64 * 136];    // -k_cum
  __shared__ bf16 SKDT[128 * 72];   // (k*exp(g_last-gc))^T [d][t]
  __shared__ bf16 SVNT[128 * 72];   // v_new^T [d][t]
  __shared__ bf16 SSTT[128 * 136];  // state^T [dv][dk]
  __shared__ bf16 SATT[64 * 72];    // attn [i][j]
  __shared__ float GCL[64];

  const int h = blockIdx.x, hk = h >> 1;
  const int tid = threadIdx.x, wave = tid >> 5, lane = tid & 31;
  const int half = lane >> 4, l16 = lane & 15;

  v8f sacc[8];
#pragma unroll
  for (int j = 0; j < 8; ++j) sacc[j] = v8f_zero();

  for (int n = 0; n < 64; ++n) {
    const int s0 = n * 64;
    // ---- phase 1: publish state^T (bf16), stage chunk inputs
#pragma unroll
    for (int j = 0; j < 8; ++j) {
      int col = j * 16 + l16;  // dv
#pragma unroll
      for (int i8 = 0; i8 < 8; ++i8) {
        int m = wave * 16 + i8 + 8 * half;  // dk
        SSTT[(size_t)col * 136 + m] = (bf16)sacc[j][i8];
      }
    }
    {
      const int r = tid >> 2, c0 = (tid & 3) * 32;
      const bf16* qrow = q_bf + ((size_t)(s0 + r) * 16 + hk) * 128;
      const bf16* krow = k_bf + ((size_t)(s0 + r) * 16 + hk) * 128;
      const bf16* kcrow = kc_bf + (((size_t)h * 64 + n) * 64 + r) * 128;
      for (int c = c0; c < c0 + 32; ++c) {
        SQ[r * 136 + c] = qrow[c];
        SKK[r * 136 + c] = krow[c];
        SKC[r * 136 + c] = (bf16)(-(float)kcrow[c]);
      }
      if (tid < 64) GCL[tid] = gc[(size_t)(s0 + tid) * 32 + h];
    }
    __syncthreads();

    // ---- phase 2: v_new = v_t - k_cum@state -> SVNT ; attn -> SATT
    {
      const int mt = wave >> 1;
      const bf16* vtbase = vt_bf + ((size_t)h * 64 + n) * 64 * 128;
#pragma unroll
      for (int jj = 0; jj < 4; ++jj) {
        int nt = (wave & 1) * 4 + jj;
        v8f c;
        int d = nt * 16 + l16;
#pragma unroll
        for (int i8 = 0; i8 < 8; ++i8) {
          int t = mt * 16 + i8 + 8 * half;
          c[i8] = (float)vtbase[(size_t)t * 128 + d];
        }
#pragma unroll
        for (int kk = 0; kk < 128; kk += 32) {
          v16bf a = frag_a_lds(SKC, 136, mt * 16, kk);
          v16bf b = frag_b_lds(SSTT, 136, nt * 16, kk);
          c = wmma_bf16(a, b, c);
        }
#pragma unroll
        for (int i8 = 0; i8 < 8; ++i8) {
          int t = mt * 16 + i8 + 8 * half;
          SVNT[(size_t)(nt * 16 + l16) * 72 + t] = (bf16)c[i8];
        }
      }
#pragma unroll
      for (int jj = 0; jj < 2; ++jj) {
        int nt = (wave & 1) * 2 + jj;
        v8f c = v8f_zero();
#pragma unroll
        for (int kk = 0; kk < 128; kk += 32) {
          v16bf a = frag_a_lds(SQ, 136, mt * 16, kk);
          v16bf b = frag_b_lds(SKK, 136, nt * 16, kk);
          c = wmma_bf16(a, b, c);
        }
        int j = nt * 16 + l16;
#pragma unroll
        for (int i8 = 0; i8 < 8; ++i8) {
          int i = mt * 16 + i8 + 8 * half;
          float v = (i >= j) ? c[i8] * __expf(GCL[i] - GCL[j]) : 0.0f;
          SATT[i * 72 + j] = (bf16)v;
        }
      }
    }
    __syncthreads();

    // ---- phase 3: SQ *= exp(gc) in place ; build k_dec^T
    {
      const int r = tid >> 2, c0 = (tid & 3) * 32;
      const float eg = __expf(GCL[r]);
      for (int c = c0; c < c0 + 32; ++c)
        SQ[r * 136 + c] = (bf16)((float)SQ[r * 136 + c] * eg);
      const float gl = GCL[63];
      const int d = tid >> 1, t0 = (tid & 1) * 32;
      for (int t = t0; t < t0 + 32; ++t)
        SKDT[d * 72 + t] = (bf16)((float)SKK[t * 136 + d] * __expf(gl - GCL[t]));
    }
    __syncthreads();

    // ---- phase 4: out = qe@state + attn@v_new ; state = state*exp(gl) + kdT@v_new
    {
      const int mt = wave >> 1;
#pragma unroll
      for (int jj = 0; jj < 4; ++jj) {
        int nt = (wave & 1) * 4 + jj;
        v8f c = v8f_zero();
#pragma unroll
        for (int kk = 0; kk < 128; kk += 32) {
          v16bf a = frag_a_lds(SQ, 136, mt * 16, kk);
          v16bf b = frag_b_lds(SSTT, 136, nt * 16, kk);
          c = wmma_bf16(a, b, c);
        }
#pragma unroll
        for (int kk = 0; kk < 64; kk += 32) {
          v16bf a = frag_a_lds(SATT, 72, mt * 16, kk);
          v16bf b = frag_b_lds(SVNT, 72, nt * 16, kk);
          c = wmma_bf16(a, b, c);
        }
        int d = nt * 16 + l16;
#pragma unroll
        for (int i8 = 0; i8 < 8; ++i8) {
          int t = mt * 16 + i8 + 8 * half;
          core[((size_t)(s0 + t) * 32 + h) * 128 + d] = c[i8];
        }
      }
      const float egl = __expf(GCL[63]);
#pragma unroll
      for (int j = 0; j < 8; ++j) {
#pragma unroll
        for (int i8 = 0; i8 < 8; ++i8) sacc[j][i8] *= egl;
#pragma unroll
        for (int kk = 0; kk < 64; kk += 32) {
          v16bf a = frag_a_lds(SKDT, 72, wave * 16, kk);
          v16bf b = frag_b_lds(SVNT, 72, j * 16, kk);
          sacc[j] = wmma_bf16(a, b, sacc[j]);
        }
      }
    }
    __syncthreads();
  }
}

// ---------------------------------------------------------------------------
// Gated RMSNorm: y = (w * rmsnorm(core)) * silu(z). 1 wave per (s,h).
// ---------------------------------------------------------------------------
__global__ __launch_bounds__(256) void normgate_kernel(
    const float* __restrict__ core, const float* __restrict__ z_raw,
    const float* __restrict__ w, float* __restrict__ y) {
  const int wave = threadIdx.x >> 5, lane = threadIdx.x & 31;
  const int task = blockIdx.x * 8 + wave;
  const int s = task >> 5, h = task & 31;
  const float* x = core + ((size_t)s * 32 + h) * 128;
  const float* z = z_raw + (size_t)s * 4096 + h * 128;
  float v[4];
  float ss = 0.0f;
#pragma unroll
  for (int i = 0; i < 4; ++i) { v[i] = x[lane + 32 * i]; ss += v[i] * v[i]; }
#pragma unroll
  for (int m = 16; m >= 1; m >>= 1) ss += __shfl_xor(ss, m, 32);
  const float r = rsqrtf(ss * (1.0f / 128.0f) + 1e-6f);
  float* yo = y + (size_t)s * 4096 + h * 128;
#pragma unroll
  for (int i = 0; i < 4; ++i) {
    int d = lane + 32 * i;
    yo[d] = w[d] * (v[i] * r) * siluf_(z[d]);
  }
}

// ---------------------------------------------------------------------------
// Host launcher
// ---------------------------------------------------------------------------
extern "C" void kernel_launch(void* const* d_in, const int* in_sizes, int n_in,
                              void* d_out, int out_size, void* d_ws, size_t ws_size,
                              hipStream_t stream) {
  (void)in_sizes; (void)n_in; (void)out_size; (void)ws_size;
  const int S = 4096, HD = 2048, CD = 8192, VD = 4096;

  const float* hidden = (const float*)d_in[0];
  const float* W_qkv  = (const float*)d_in[1];
  const float* W_z    = (const float*)d_in[2];
  const float* W_b    = (const float*)d_in[3];
  const float* W_a    = (const float*)d_in[4];
  const float* conv_w = (const float*)d_in[5];
  const float* dt_bias = (const float*)d_in[6];
  const float* A_log   = (const float*)d_in[7];
  const float* norm_w  = (const float*)d_in[8];
  const float* W_out   = (const float*)d_in[9];

  const size_t MB = 1ull << 20;
  char* ws = (char*)d_ws;
  float* qkv_raw = (float*)(ws + 0);          // 128 MB (reused later)
  float* qkv_act = (float*)(ws + 128 * MB);   // 128 MB
  float* z_raw   = (float*)(ws + 256 * MB);   // 64 MB
  float* beta_raw = (float*)(ws + 320 * MB);
  float* a_raw    = (float*)(ws + 321 * MB);
  float* g_buf    = (float*)(ws + 322 * MB);
  float* gc_buf   = (float*)(ws + 323 * MB);
  float* beta_buf = (float*)(ws + 324 * MB);
  bf16* q_bf  = (bf16*)(ws + 325 * MB);       // 16 MB
  bf16* k_bf  = (bf16*)(ws + 341 * MB);       // 16 MB
  bf16* v_bf  = (bf16*)(ws + 357 * MB);       // 32 MB
  bf16* vt_bf = (bf16*)(ws + 389 * MB);       // 32 MB
  bf16* kc_bf = (bf16*)(ws + 421 * MB);       // 32 MB  (total 453 MB)
  float* core = (float*)(ws + 0);             // reuse qkv_raw region (64 MB)
  float* ybuf = (float*)(ws + 64 * MB);       // reuse (64 MB)

  dim3 blk(256);

  // projections
  gemm_bf16_kernel<<<dim3(CD / 128, S / 128), blk, 0, stream>>>(hidden, W_qkv, qkv_raw, S, CD, HD);
  gemm_bf16_kernel<<<dim3(VD / 128, S / 128), blk, 0, stream>>>(hidden, W_z, z_raw, S, VD, HD);
  gemm_bf16_kernel<<<dim3(1, S / 128), blk, 0, stream>>>(hidden, W_b, beta_raw, S, 32, HD);
  gemm_bf16_kernel<<<dim3(1, S / 128), blk, 0, stream>>>(hidden, W_a, a_raw, S, 32, HD);

  // conv + activations + packs
  conv_silu_kernel<<<dim3(CD / 256, S), blk, 0, stream>>>(qkv_raw, conv_w, qkv_act, S, CD);
  qknorm_kernel<<<dim3(S * 32 / 8), blk, 0, stream>>>(qkv_act, q_bf, k_bf);
  vpack_kernel<<<dim3(S * VD / 256), blk, 0, stream>>>(qkv_act, v_bf);
  betag_kernel<<<dim3(S * 32 / 256), blk, 0, stream>>>(beta_raw, a_raw, dt_bias, A_log, beta_buf, g_buf);
  gcum_kernel<<<dim3(64 * 32 / 256), blk, 0, stream>>>(g_buf, gc_buf);

  // chunked gated delta rule
  chunk_prep_kernel<<<dim3(64, 32), blk, 0, stream>>>(k_bf, v_bf, beta_buf, gc_buf, vt_bf, kc_bf);
  scan_kernel<<<dim3(32), blk, 0, stream>>>(q_bf, k_bf, gc_buf, vt_bf, kc_bf, core);

  // norm + output projection
  normgate_kernel<<<dim3(S * 32 / 8), blk, 0, stream>>>(core, z_raw, norm_w, ybuf);
  gemm_bf16_kernel<<<dim3(HD / 128, S / 128), blk, 0, stream>>>(ybuf, W_out, (float*)d_out, S, HD, VD);
}
